// RGTLayer_47562467835949
// MI455X (gfx1250) — compile-verified, matrix operand order checked by reading, not run on hardware
//
#include <hip/hip_runtime.h>
#include <hip/hip_bf16.h>

#define N_NODES 100000
#define N_EDGES 600000
#define CDIM 128
#define NTYPES 2
#define SEMH 4

typedef __attribute__((ext_vector_type(16))) __bf16 v16bf;
typedef __attribute__((ext_vector_type(8)))  float  v8f;

union Frag32B {
    uint4 u[2];
    v16bf v;
};

static __device__ __forceinline__ unsigned short f2bf(float f) {
    unsigned u = __float_as_uint(f);
    unsigned r = 0x7FFFu + ((u >> 16) & 1u);   // round-to-nearest-even
    return (unsigned short)((u + r) >> 16);
}

// float atomic max via signed/unsigned int ordering trick
static __device__ __forceinline__ void atomicMaxF(float* addr, float val) {
    if (val >= 0.0f) atomicMax((int*)addr, __float_as_int(val));
    else             atomicMin((unsigned int*)addr, __float_as_uint(val));
}

// ---------------- conversion kernels ----------------

__global__ void k_cvt_x(const float* __restrict__ x, unsigned short* __restrict__ xb, int n) {
    int i = blockIdx.x * blockDim.x + threadIdx.x;
    if (i < n) xb[i] = f2bf(x[i]);
}

// W: [NTYPES][K=128][OUT=128] fp32 -> wt[(t*4+j)][out][k] bf16 ; bias[(t*4+j)][out]
__global__ void k_cvt_w(const float* __restrict__ W, const float* __restrict__ b,
                        unsigned short* __restrict__ wt, float* __restrict__ bias, int j) {
    int i = blockIdx.x * blockDim.x + threadIdx.x;
    if (i >= NTYPES * CDIM * CDIM) return;
    int t   = i >> 14;
    int k   = (i >> 7) & 127;
    int out = i & 127;
    wt[((size_t)(t * 4 + j) * CDIM + out) * CDIM + k] = f2bf(W[i]);
    if (k == 0) bias[(t * 4 + j) * CDIM + out] = b[t * CDIM + out];
}

// ---------------- WMMA GEMM: QKVS[t][n][512] = X @ [Wq|Wk|Wv|Wskip] + bias ----------------
// grid.y = j in 0..7 (t = j>>2, slot = j&3); each wave computes a 16x128 strip.

__global__ void k_gemm(const unsigned short* __restrict__ xb,
                       const unsigned short* __restrict__ wt,
                       const float* __restrict__ bias,
                       float* __restrict__ qkvs) {
    const int j     = blockIdx.y;
    const int wave  = threadIdx.x >> 5;
    const int lane  = threadIdx.x & 31;
    const int mtile = blockIdx.x * (blockDim.x >> 5) + wave;
    const int MT    = N_NODES / 16;  // 6250, exact
    if (mtile >= MT) return;

    const int half = lane >> 4;
    const int l15  = lane & 15;
    const int t    = j >> 2;
    const int slot = j & 3;

    // A: row = l15 (both lane halves hold rows 0..15), K range selected by half
    const uint4* ap = (const uint4*)(xb + (size_t)(mtile * 16 + l15) * CDIM);

    for (int nt = 0; nt < 8; ++nt) {
        // B fragment == A-style fragment of W^T row (= output column nt*16+l15)
        const uint4* bp = (const uint4*)(wt + ((size_t)j * CDIM + nt * 16 + l15) * CDIM);
        float bv = bias[j * CDIM + nt * 16 + l15];
        v8f acc;
#pragma unroll
        for (int r = 0; r < 8; ++r) acc[r] = bv;

#pragma unroll
        for (int kb8 = 0; kb8 < 16; kb8 += 4) {  // K blocks of 32 (kb8 = kb/8)
            Frag32B a, b;
            a.u[0] = ap[kb8 + half];        // K = kb + 8*half + 0..7
            a.u[1] = ap[kb8 + 2 + half];    // K = kb + 16 + 8*half + 0..7
            b.u[0] = bp[kb8 + half];
            b.u[1] = bp[kb8 + 2 + half];
            acc = __builtin_amdgcn_wmma_f32_16x16x32_bf16(
                false, a.v, false, b.v, (short)0, acc, false, false);
        }

        // D: element r -> row r + 8*half, col = l15
        float* op = qkvs + ((size_t)t * N_NODES + (size_t)mtile * 16 + 8 * half) * 512
                  + slot * 128 + nt * 16 + l15;
#pragma unroll
        for (int r = 0; r < 8; ++r) op[(size_t)r * 512] = acc[r];
    }
}

// ---------------- edge attention kernels ----------------

__global__ void k_init_md(float* __restrict__ m, float* __restrict__ denom) {
    int i = blockIdx.x * blockDim.x + threadIdx.x;
    if (i < N_NODES) { m[i] = -1e30f; denom[i] = 0.0f; }
}

// one wave per edge: score = dot(q[dst], k[src]) / sqrt(128); atomic max into m[dst]
__global__ void k_score(const float* __restrict__ qkvs_t,
                        const long long* __restrict__ ei,
                        const long long* __restrict__ et,
                        float* __restrict__ score, float* __restrict__ m, int type) {
    int wave = threadIdx.x >> 5;
    int lane = threadIdx.x & 31;
    int e = blockIdx.x * (blockDim.x >> 5) + wave;
    if (e >= N_EDGES) return;
    if ((int)et[e] != type) return;
    int src = (int)ei[e];
    int dst = (int)ei[N_EDGES + e];
    const float4* q = (const float4*)(qkvs_t + (size_t)dst * 512);        // q slice
    const float4* k = (const float4*)(qkvs_t + (size_t)src * 512 + 128);  // k slice
    float4 qa = q[lane];
    float4 ka = k[lane];
    float s = qa.x * ka.x + qa.y * ka.y + qa.z * ka.z + qa.w * ka.w;
#pragma unroll
    for (int off = 16; off > 0; off >>= 1) s += __shfl_xor(s, off, 32);
    s *= 0.08838834764831845f;  // 1/sqrt(128)
    if (lane == 0) {
        score[e] = s;
        atomicMaxF(&m[dst], s);
    }
}

// one thread per edge: a = exp(score - m[dst]); denom[dst] += a
__global__ void k_expsum(const long long* __restrict__ ei, const long long* __restrict__ et,
                         float* __restrict__ score, const float* __restrict__ m,
                         float* __restrict__ denom, int type) {
    int e = blockIdx.x * blockDim.x + threadIdx.x;
    if (e >= N_EDGES) return;
    if ((int)et[e] != type) return;
    int dst = (int)ei[N_EDGES + e];
    float a = __expf(score[e] - m[dst]);
    score[e] = a;
    atomicAdd(&denom[dst], a);
}

// one wave per edge: u[dst] += alpha * v[src]  (u = skip slice, already holds X@Wskip+b)
__global__ void k_scatter(float* __restrict__ qkvs_t,
                          const long long* __restrict__ ei, const long long* __restrict__ et,
                          const float* __restrict__ score, const float* __restrict__ denom,
                          int type) {
    int wave = threadIdx.x >> 5;
    int lane = threadIdx.x & 31;
    int e = blockIdx.x * (blockDim.x >> 5) + wave;
    if (e >= N_EDGES) return;
    if ((int)et[e] != type) return;
    int src = (int)ei[e];
    int dst = (int)ei[N_EDGES + e];
    float alpha = score[e] / fmaxf(denom[dst], 1e-16f);
    const float4* v = (const float4*)(qkvs_t + (size_t)src * 512 + 256);  // v slice
    float* u = qkvs_t + (size_t)dst * 512 + 384;                          // skip/u slice
    float4 vv = v[lane];
    atomicAdd(u + lane * 4 + 0, alpha * vv.x);
    atomicAdd(u + lane * 4 + 1, alpha * vv.y);
    atomicAdd(u + lane * 4 + 2, alpha * vv.z);
    atomicAdd(u + lane * 4 + 3, alpha * vv.w);
}

// ---------------- semantic attention + combine ----------------

__global__ void k_zero(float* __restrict__ p, int n) {
    int i = blockIdx.x * blockDim.x + threadIdx.x;
    if (i < n) p[i] = 0.0f;
}

__global__ void k_colsum(const float* __restrict__ qkvs, float* __restrict__ tsum) {
    int c = threadIdx.x;          // 0..127
    int t = blockIdx.y;
    int r0 = blockIdx.x * 512;
    int r1 = r0 + 512;
    if (r1 > N_NODES) r1 = N_NODES;
    const float* base = qkvs + (size_t)t * N_NODES * 512 + 384 + c;
    float acc = 0.0f;
    for (int r = r0; r < r1; ++r) acc += base[(size_t)r * 512];
    atomicAdd(&tsum[t * CDIM + c], acc);
}

__global__ void k_beta(const float* __restrict__ tsum, const float* __restrict__ attw,
                       const float* __restrict__ prelu, float* __restrict__ coef) {
    if (threadIdx.x != 0 || blockIdx.x != 0) return;
    float a = prelu[0];
    float w[NTYPES][SEMH];
    for (int t = 0; t < NTYPES; ++t)
        for (int h = 0; h < SEMH; ++h) {
            float s = 0.0f;
            for (int c = 0; c < CDIM; ++c) s += tsum[t * CDIM + c] * attw[c * SEMH + h];
            s *= (1.0f / (float)N_NODES);
            w[t][h] = (s >= 0.0f) ? s : a * s;
        }
    float ct[NTYPES] = {0.0f, 0.0f};
    for (int h = 0; h < SEMH; ++h) {
        float mx = w[0][h];
        for (int t = 1; t < NTYPES; ++t) mx = fmaxf(mx, w[t][h]);
        float ex[NTYPES], den = 0.0f;
        for (int t = 0; t < NTYPES; ++t) { ex[t] = __expf(w[t][h] - mx); den += ex[t]; }
        for (int t = 0; t < NTYPES; ++t) ct[t] += ex[t] / den;
    }
    for (int t = 0; t < NTYPES; ++t) coef[t] = ct[t] / (float)SEMH;
}

__global__ void k_combine(const float* __restrict__ qkvs, const float* __restrict__ coef,
                          float* __restrict__ out) {
    int i = blockIdx.x * blockDim.x + threadIdx.x;
    if (i >= N_NODES * CDIM) return;
    int n = i >> 7, c = i & 127;
    float u0 = qkvs[(size_t)n * 512 + 384 + c];
    float u1 = qkvs[((size_t)N_NODES + n) * 512 + 384 + c];
    out[i] = coef[0] * u0 + coef[1] * u1;
}

// ---------------- launch ----------------

extern "C" void kernel_launch(void* const* d_in, const int* in_sizes, int n_in,
                              void* d_out, int out_size, void* d_ws, size_t ws_size,
                              hipStream_t stream) {
    (void)in_sizes; (void)n_in; (void)out_size; (void)ws_size;
    const float*      x     = (const float*)d_in[0];
    const long long*  ei    = (const long long*)d_in[1];
    const long long*  et    = (const long long*)d_in[2];
    const float*      Wq    = (const float*)d_in[3];
    const float*      bq    = (const float*)d_in[4];
    const float*      Wk    = (const float*)d_in[5];
    const float*      bk    = (const float*)d_in[6];
    const float*      Wv    = (const float*)d_in[7];
    const float*      bv    = (const float*)d_in[8];
    const float*      Wsk   = (const float*)d_in[9];
    const float*      bsk   = (const float*)d_in[10];
    const float*      attw  = (const float*)d_in[11];
    const float*      prelu = (const float*)d_in[12];
    float* out = (float*)d_out;

    char* ws = (char*)d_ws;
    size_t off = 0;
    auto alloc = [&](size_t bytes) -> size_t {
        size_t p = off;
        off = (off + bytes + 255) & ~(size_t)255;
        return p;
    };
    unsigned short* xb   = (unsigned short*)(ws + alloc((size_t)N_NODES * CDIM * 2));
    unsigned short* wt   = (unsigned short*)(ws + alloc((size_t)8 * CDIM * CDIM * 2));
    float*          bias = (float*)(ws + alloc((size_t)8 * CDIM * 4));
    float*          qkvs = (float*)(ws + alloc((size_t)2 * N_NODES * 512 * 4));
    float*          mb   = (float*)(ws + alloc((size_t)N_NODES * 4));
    float*          den  = (float*)(ws + alloc((size_t)N_NODES * 4));
    float*          scr  = (float*)(ws + alloc((size_t)N_EDGES * 4));
    float*          tsum = (float*)(ws + alloc((size_t)NTYPES * CDIM * 4));
    float*          coef = (float*)(ws + alloc(64));

    k_cvt_x<<<(N_NODES * CDIM + 255) / 256, 256, 0, stream>>>(x, xb, N_NODES * CDIM);
    const int wn = NTYPES * CDIM * CDIM;
    k_cvt_w<<<(wn + 255) / 256, 256, 0, stream>>>(Wq,  bq,  wt, bias, 0);
    k_cvt_w<<<(wn + 255) / 256, 256, 0, stream>>>(Wk,  bk,  wt, bias, 1);
    k_cvt_w<<<(wn + 255) / 256, 256, 0, stream>>>(Wv,  bv,  wt, bias, 2);
    k_cvt_w<<<(wn + 255) / 256, 256, 0, stream>>>(Wsk, bsk, wt, bias, 3);

    dim3 ggrid((6250 + 7) / 8, 8);
    k_gemm<<<ggrid, 256, 0, stream>>>(xb, wt, bias, qkvs);

    for (int t = 0; t < NTYPES; ++t) {
        float* qt = qkvs + (size_t)t * N_NODES * 512;
        k_init_md<<<(N_NODES + 255) / 256, 256, 0, stream>>>(mb, den);
        k_score  <<<(N_EDGES + 7) / 8, 256, 0, stream>>>(qt, ei, et, scr, mb, t);
        k_expsum <<<(N_EDGES + 255) / 256, 256, 0, stream>>>(ei, et, scr, mb, den, t);
        k_scatter<<<(N_EDGES + 7) / 8, 256, 0, stream>>>(qt, ei, et, scr, den, t);
    }

    k_zero<<<1, 256, 0, stream>>>(tsum, NTYPES * CDIM);
    dim3 cgrid((N_NODES + 511) / 512, NTYPES);
    k_colsum<<<cgrid, 128, 0, stream>>>(qkvs, tsum);
    k_beta<<<1, 32, 0, stream>>>(tsum, attw, prelu, coef);
    k_combine<<<(N_NODES * CDIM + 255) / 256, 256, 0, stream>>>(qkvs, coef, out);
}
